// GNNAutoDecoder_55216099558068
// MI455X (gfx1250) — compile-verified, matrix operand order checked by reading, not run on hardware
//
#include <hip/hip_runtime.h>
#include <stdint.h>

// ---------------------------------------------------------------------------
// NWR-GAE fused decoder loss for MI455X (gfx1250, wave32, WMMA 16x16x32 f16)
// N=20000 nodes, S=10 samples, H=IN=128, MAXDEG=32
// One block = 16 nodes, 256 threads = 8 waves; wave w owns output cols [16w,16w+16)
// All GEMMs on v_wmma_f32_16x16x32_f16; full dataflow stays in LDS/VGPRs.
// ---------------------------------------------------------------------------

#define N_NODES 20000
#define S_SAMP  10
#define HDIM    128
#define IN_DIM  128
#define MAXDEG  32
#define NT      16                 // nodes per block
#define NBLK    (N_NODES / NT)     // 1250
#define THREADS 256
#define LAMBDA1 1.0e-2f
#define LAMBDA2 1.0f

typedef __attribute__((ext_vector_type(16))) _Float16 v16h;
typedef __attribute__((ext_vector_type(8)))  float    v8f;

union HFrag {
  v16h     v;
  _Float16 e[16];
  float4   q[2];
};
union PackH2 { _Float16 h[2]; uint32_t u; };
union PackH4 { _Float16 h[4]; float2 f2; };

// ---- A-fragment (16x32 f16) from LDS row-major [16][128] tile.
// ISA 7.12.2 16-bit A layout: lane half selects K-interleave:
//   elem j<8  -> K = kbase + half*8 + j
//   elem j>=8 -> K = kbase + 16 + half*8 + (j-8)
// Both 8-element runs are contiguous -> two b128 LDS loads.
__device__ __forceinline__ v16h load_a_lds(const _Float16* act, int lane, int kbase) {
  const int m    = lane & 15;
  const int half = lane >> 4;
  HFrag f;
  f.q[0] = *(const float4*)(act + m * HDIM + kbase + half * 8);
  f.q[1] = *(const float4*)(act + m * HDIM + kbase + 16 + half * 8);
  return f.v;
}

// ---- B-fragment (32x16 f16) from global f32 weight W[k][n] (row-major 128x128),
// converting on the fly. Lane holds column n = ncol, K striped like A.
__device__ __forceinline__ v16h load_b_global(const float* __restrict__ W,
                                              int lane, int kbase, int ncol) {
  const int half = lane >> 4;
  HFrag f;
#pragma unroll
  for (int j = 0; j < 8; ++j) {
    const int k0 = kbase + half * 8 + j;
    const int k1 = kbase + 16 + half * 8 + j;
    f.e[j]     = (_Float16)W[k0 * HDIM + ncol];
    f.e[j + 8] = (_Float16)W[k1 * HDIM + ncol];
  }
  return f.v;
}

__device__ __forceinline__ v8f wmma16(v16h a, v16h b, v8f c) {
  // D = A(16x32) * B(32x16) + C, f32 accumulate
  return __builtin_amdgcn_wmma_f32_16x16x32_f16(false, a, false, b, (short)0, c,
                                                false, false);
}

// ---- C tile (16x16 f32) -> LDS f16 activation tile, +bias, optional ReLU.
// C layout: VGPR r, lane L: row = r + 8*(L>>4), col = L&15 (+ wave offset in ncol).
__device__ __forceinline__ void store_c_act(_Float16* act, v8f c, float bias,
                                            int lane, int ncol, bool relu) {
  const int half = lane >> 4;
#pragma unroll
  for (int r = 0; r < 8; ++r) {
    float v = c[r] + bias;
    if (relu) v = fmaxf(v, 0.0f);
    act[(r + 8 * half) * HDIM + ncol] = (_Float16)v;
  }
}

// ---- counter-based RNG (deterministic; JAX threefry not bit-matchable) ----
__device__ __forceinline__ unsigned hmix(unsigned x) {
  x ^= x >> 16; x *= 0x7feb352dU;
  x ^= x >> 15; x *= 0x846ca68bU;
  x ^= x >> 16;
  return x;
}
__device__ __forceinline__ unsigned hash3(unsigned a, unsigned b, unsigned c) {
  return hmix(a * 0x9E3779B1u + b * 0x85EBCA77u + c * 0xC2B2AE3Du + 0x27220A95u);
}
__device__ __forceinline__ float u01(unsigned x) {
  return ((float)(x >> 8) + 0.5f) * (1.0f / 16777216.0f);
}

struct SMem {                       // 63232 bytes (< 64 KB static LDS)
  float    meanL[NT * HDIM];        // 8 KB
  float    sigL[NT * HDIM];         // 8 KB
  _Float16 act[NT * HDIM];          // 4 KB  (h2 / MLP activations / var / gen)
  _Float16 tgtb[NT * S_SAMP * HDIM];// 40 KB (gathered neighbor targets, f16)
  float    t2[NT * S_SAMP];         // target squared norms
  int      degL[NT];
  int      mlenL[NT];
  float    red[THREADS];            // block reduction scratch
};

__device__ __forceinline__ float blockReduceSum(float v, float* red) {
  const int tid = threadIdx.x;
  red[tid] = v;
  __syncthreads();
  for (int s = THREADS / 2; s > 0; s >>= 1) {
    if (tid < s) red[tid] += red[tid + s];
    __syncthreads();
  }
  const float r = red[0];
  __syncthreads();
  return r;
}

__global__ __launch_bounds__(THREADS)
void nwr_gae_fused(const int* __restrict__ degree, const int* __restrict__ neighbors,
                   const float* __restrict__ h0, const float* __restrict__ h1,
                   const float* __restrict__ h2,
                   const float* __restrict__ fdW1, const float* __restrict__ fdb1,
                   const float* __restrict__ fdW2, const float* __restrict__ fdb2,
                   const float* __restrict__ fdW3, const float* __restrict__ fdb3,
                   const float* __restrict__ gW1,  const float* __restrict__ gb1,
                   const float* __restrict__ gW2,  const float* __restrict__ gb2,
                   const float* __restrict__ gW3,  const float* __restrict__ gb3,
                   const float* __restrict__ mW,   const float* __restrict__ mb,
                   const float* __restrict__ sW,   const float* __restrict__ sb,
                   float* __restrict__ accum) {
  __shared__ SMem sm;
  const int tid  = threadIdx.x;
  const int lane = tid & 31;
  const int wave = tid >> 5;
  const int half = lane >> 4;
  const int n0   = blockIdx.x * NT;
  const int ncol = wave * 16 + (lane & 15);   // this thread's output column

  // Warm L2/WGP$ with generator weights (global_prefetch_b8), reused 30x/block.
  __builtin_prefetch(gW1 + tid * 64, 0, 0);
  __builtin_prefetch(gW2 + tid * 64, 0, 0);
  __builtin_prefetch(gW3 + tid * 64, 0, 0);

  // ---- stage h2 tile as f16 (float4 loads, packed 8B LDS stores) ----
  for (int v = tid; v < NT * (HDIM / 4); v += THREADS) {
    const int r = v / (HDIM / 4);
    const int c4 = (v - r * (HDIM / 4)) * 4;
    const float4 x = *(const float4*)(h2 + (size_t)(n0 + r) * HDIM + c4);
    PackH4 pk;
    pk.h[0] = (_Float16)x.x; pk.h[1] = (_Float16)x.y;
    pk.h[2] = (_Float16)x.z; pk.h[3] = (_Float16)x.w;
    *(float2*)(&sm.act[r * HDIM + c4]) = pk.f2;
  }
  if (tid < NT) {
    const int d = degree[n0 + tid];
    sm.degL[tid]  = d;
    sm.mlenL[tid] = d < S_SAMP ? d : S_SAMP;
  }
  __syncthreads();

  // ================= Phase 1: mean / sigma / feature decoder =================
  v16h a0[4];
#pragma unroll
  for (int kc = 0; kc < 4; ++kc) a0[kc] = load_a_lds(sm.act, lane, kc * 32);

  v8f cm = {}, cs = {}, c1 = {};
#pragma unroll
  for (int kc = 0; kc < 4; ++kc) {
    cm = wmma16(a0[kc], load_b_global(mW,   lane, kc * 32, ncol), cm);
    cs = wmma16(a0[kc], load_b_global(sW,   lane, kc * 32, ncol), cs);
    c1 = wmma16(a0[kc], load_b_global(fdW1, lane, kc * 32, ncol), c1);
  }
  {
    const float bm = mb[ncol], bs = sb[ncol];
#pragma unroll
    for (int r = 0; r < 8; ++r) {
      const int row = (r + 8 * half) * HDIM + ncol;
      sm.meanL[row] = cm[r] + bm;
      sm.sigL[row]  = expf(cs[r] + bs);
    }
  }
  __syncthreads();                                    // everyone done reading h2
  store_c_act(sm.act, c1, fdb1[ncol], lane, ncol, true);
  __syncthreads();

  {                                                   // fd layer 2
    v16h a[4];
#pragma unroll
    for (int kc = 0; kc < 4; ++kc) a[kc] = load_a_lds(sm.act, lane, kc * 32);
    __syncthreads();
    v8f c = {};
#pragma unroll
    for (int kc = 0; kc < 4; ++kc)
      c = wmma16(a[kc], load_b_global(fdW2, lane, kc * 32, ncol), c);
    store_c_act(sm.act, c, fdb2[ncol], lane, ncol, true);
    __syncthreads();
  }
  float fpart = 0.0f;
  {                                                   // fd layer 3 + MSE partial
    v16h a[4];
#pragma unroll
    for (int kc = 0; kc < 4; ++kc) a[kc] = load_a_lds(sm.act, lane, kc * 32);
    v8f c = {};
#pragma unroll
    for (int kc = 0; kc < 4; ++kc)
      c = wmma16(a[kc], load_b_global(fdW3, lane, kc * 32, ncol), c);
    const float b3 = fdb3[ncol];
#pragma unroll
    for (int r = 0; r < 8; ++r) {
      const int m = r + 8 * half;
      const float d = (c[r] + b3) - h0[(size_t)(n0 + m) * HDIM + ncol];
      fpart += d * d;
    }
  }
  const float fsum = blockReduceSum(fpart, sm.red);
  if (tid == 0) atomicAdd(&accum[0], fsum);

  // ================= Phase 2: generator + chamfer matching ==================
  // Wave-resident B-fragments of the generator weights (96 VGPRs/lane).
  v16h gwf[3][4];
#pragma unroll
  for (int kc = 0; kc < 4; ++kc) {
    gwf[0][kc] = load_b_global(gW1, lane, kc * 32, ncol);
    gwf[1][kc] = load_b_global(gW2, lane, kc * 32, ncol);
    gwf[2][kc] = load_b_global(gW3, lane, kc * 32, ncol);
  }
  const float gbias[3] = {gb1[ncol], gb2[ncol], gb3[ncol]};

  float hAccum = 0.0f;

  for (int rep = 0; rep < 3; ++rep) {
    __syncthreads();
    // ---- gather targets: tgt[n,s,:] = h1[sampled neighbor] (zero-padded) ----
    // float4 global loads (coalesced, L2-resident h1) + packed 8B LDS stores.
    for (int v = tid; v < NT * S_SAMP * (HDIM / 4); v += THREADS) {
      const int row = v / (HDIM / 4);
      const int c4  = (v - row * (HDIM / 4)) * 4;
      const int nl = row / S_SAMP, s = row - nl * S_SAMP;
      const int n  = n0 + nl;
      float4 val = make_float4(0.0f, 0.0f, 0.0f, 0.0f);
      const int deg = sm.degL[nl];
      if (s < sm.mlenL[nl]) {
        const float u = u01(hash3(0xB0B0u + (unsigned)rep, (unsigned)n, (unsigned)s));
        int rp = (int)(u * (float)deg);
        if (rp >= deg) rp = deg - 1;
        const int sq  = (s < deg - 1) ? s : (deg - 1);
        const int pos = (deg < S_SAMP) ? sq : rp;
        const int idx = neighbors[(size_t)n * MAXDEG + pos];
        val = *(const float4*)(h1 + (size_t)idx * HDIM + c4);
      }
      PackH4 pk;
      pk.h[0] = (_Float16)val.x; pk.h[1] = (_Float16)val.y;
      pk.h[2] = (_Float16)val.z; pk.h[3] = (_Float16)val.w;
      *(float2*)(&sm.tgtb[row * HDIM + c4]) = pk.f2;
    }
    __syncthreads();
    if (tid < NT * S_SAMP) {                          // target squared norms
      float s2 = 0.0f;
      for (int c2 = 0; c2 < HDIM; ++c2) {
        const float v = (float)sm.tgtb[tid * HDIM + c2];
        s2 += v * v;
      }
      sm.t2[tid] = s2;
    }
    __syncthreads();

    for (int mt = 0; mt < S_SAMP; ++mt) {             // 10 M-tiles of 16 rows
      // ---- var = mean + sig * z : one Box-Muller pair -> two normals ----
      for (int p = tid; p < NT * (HDIM / 2); p += THREADS) {
        const int i   = p / (HDIM / 2);
        const int col = (p - i * (HDIM / 2)) * 2;
        const int R = mt * 16 + i;
        const int nl = R / S_SAMP, s = R - nl * S_SAMP;
        const unsigned n = (unsigned)(n0 + nl);
        const unsigned key = (unsigned)(s * HDIM + col);
        const float u1 = fmaxf(u01(hash3(0xC001u + (unsigned)rep, n, key)), 1.0e-7f);
        const float u2 = u01(hash3(0xD00Du + (unsigned)rep, n, key));
        float sn, cn;
        __sincosf(6.28318530718f * u2, &sn, &cn);
        const float rr = sqrtf(-2.0f * __logf(u1));
        const int base = nl * HDIM + col;
        const float v0 = sm.meanL[base]     + sm.sigL[base]     * (rr * cn);
        const float v1 = sm.meanL[base + 1] + sm.sigL[base + 1] * (rr * sn);
        PackH2 pk;
        pk.h[0] = (_Float16)v0;
        pk.h[1] = (_Float16)v1;
        *(uint32_t*)(&sm.act[i * HDIM + col]) = pk.u;
      }
      __syncthreads();

      // ---- 3-layer generator MLP, all WMMA, weights in registers ----
#pragma unroll
      for (int l = 0; l < 3; ++l) {
        v16h a[4];
#pragma unroll
        for (int kc = 0; kc < 4; ++kc) a[kc] = load_a_lds(sm.act, lane, kc * 32);
        __syncthreads();
        v8f c = {};
#pragma unroll
        for (int kc = 0; kc < 4; ++kc) c = wmma16(a[kc], gwf[l][kc], c);
        store_c_act(sm.act, c, gbias[l], lane, ncol, l < 2);
        __syncthreads();
      }
      // sm.act now holds gen tile (16 rows x 128, f16)

      // ---- cost + row-min: wave handles rows {wave, wave+8} ----
#pragma unroll
      for (int ii = 0; ii < 2; ++ii) {
        const int i = wave + ii * 8;
        const int R = mt * 16 + i;
        const int nl = R / S_SAMP, s = R - nl * S_SAMP;
        const int ml = sm.mlenL[nl];
        const int hb = lane * 4;
        const float g0  = (float)sm.act[i * HDIM + hb + 0];
        const float g1v = (float)sm.act[i * HDIM + hb + 1];
        const float g2v = (float)sm.act[i * HDIM + hb + 2];
        const float g3v = (float)sm.act[i * HDIM + hb + 3];
        float gg = g0 * g0 + g1v * g1v + g2v * g2v + g3v * g3v;
        float dots[S_SAMP];
#pragma unroll
        for (int st = 0; st < S_SAMP; ++st) {
          const _Float16* tp = &sm.tgtb[(nl * S_SAMP + st) * HDIM + hb];
          dots[st] = g0 * (float)tp[0] + g1v * (float)tp[1] +
                     g2v * (float)tp[2] + g3v * (float)tp[3];
        }
#pragma unroll
        for (int off = 16; off > 0; off >>= 1) {
          gg += __shfl_xor(gg, off, 32);
#pragma unroll
          for (int st = 0; st < S_SAMP; ++st)
            dots[st] += __shfl_xor(dots[st], off, 32);
        }
        if (lane == 0 && s < ml) {
          float rmin = 1.0e30f;
#pragma unroll
          for (int st = 0; st < S_SAMP; ++st) {
            if (st < ml) {
              const float cst = gg + sm.t2[nl * S_SAMP + st] - 2.0f * dots[st];
              rmin = fminf(rmin, cst);
            }
          }
          hAccum += rmin;
        }
      }
      __syncthreads();
    }
  }

  const float hsum = blockReduceSum(hAccum, sm.red);
  if (tid == 0) atomicAdd(&accum[1], hsum);
}

__global__ void zero_accum(float* a) {
  if (threadIdx.x < 2) a[threadIdx.x] = 0.0f;
}

__global__ void finalize_loss(const float* __restrict__ a, float* __restrict__ out) {
  if (threadIdx.x == 0) {
    const float feature_loss = a[0] / (float)(N_NODES * IN_DIM);
    const float h_loss = a[1] / 3.0f;   // mean over 3 reps
    out[0] = LAMBDA1 * h_loss + LAMBDA2 * feature_loss;
  }
}

extern "C" void kernel_launch(void* const* d_in, const int* in_sizes, int n_in,
                              void* d_out, int out_size, void* d_ws, size_t ws_size,
                              hipStream_t stream) {
  (void)in_sizes; (void)n_in; (void)out_size; (void)ws_size;
  const int*   degree    = (const int*)d_in[0];
  const int*   neighbors = (const int*)d_in[1];
  const float* h0   = (const float*)d_in[2];
  const float* h1   = (const float*)d_in[3];
  const float* h2   = (const float*)d_in[4];
  const float* fdW1 = (const float*)d_in[5];
  const float* fdb1 = (const float*)d_in[6];
  const float* fdW2 = (const float*)d_in[7];
  const float* fdb2 = (const float*)d_in[8];
  const float* fdW3 = (const float*)d_in[9];
  const float* fdb3 = (const float*)d_in[10];
  const float* gW1  = (const float*)d_in[11];
  const float* gb1  = (const float*)d_in[12];
  const float* gW2  = (const float*)d_in[13];
  const float* gb2  = (const float*)d_in[14];
  const float* gW3  = (const float*)d_in[15];
  const float* gb3  = (const float*)d_in[16];
  const float* mW   = (const float*)d_in[17];
  const float* mb   = (const float*)d_in[18];
  const float* sW   = (const float*)d_in[19];
  const float* sb   = (const float*)d_in[20];

  float* accum = (float*)d_ws;
  zero_accum<<<1, 32, 0, stream>>>(accum);
  nwr_gae_fused<<<NBLK, THREADS, 0, stream>>>(
      degree, neighbors, h0, h1, h2,
      fdW1, fdb1, fdW2, fdb2, fdW3, fdb3,
      gW1, gb1, gW2, gb2, gW3, gb3,
      mW, mb, sW, sb, accum);
  finalize_loss<<<1, 32, 0, stream>>>(accum, (float*)d_out);
}
